// DGCNN_segment_57904749085241
// MI455X (gfx1250) — matrix operand order repaired and checked by reading, not compiled, not run on hardware
//
#include <hip/hip_runtime.h>

typedef __attribute__((ext_vector_type(16))) _Float16 v16h;
typedef __attribute__((ext_vector_type(8)))  float    v8f;

#define BDIM 256
#define NEG_INF (-3.4e38f)

// ---------------------------------------------------------------------------
// KNN: per wave, one row; tile columns through LDS; per-lane sorted top-20,
// wave-merge by popping max-of-heads 20 times.
// ---------------------------------------------------------------------------
#define KC 128
__global__ __launch_bounds__(BDIM) void knn_kernel(
    const float* __restrict__ feat, long sb, long sc, long sn,
    int C, int N, int* __restrict__ idxOut)
{
    __shared__ float colf[64][KC];
    __shared__ float colnorm[KC];
    __shared__ float rowf[8][64];

    const int b = blockIdx.y;
    const int tid = threadIdx.x, w = tid >> 5, lane = tid & 31;
    const int r0 = blockIdx.x * 8;

    for (int e = tid; e < 8 * 64; e += BDIM) {
        int r = e >> 6, c = e & 63;
        int cc = c < C ? c : C - 1;
        float v = feat[b * sb + (long)cc * sc + (long)(r0 + r) * sn];
        rowf[r][c] = (c < C) ? v : 0.f;
    }
    __syncthreads();

    float rn = 0.f;
    for (int c = 0; c < C; ++c) { float t = rowf[w][c]; rn += t * t; }

    float lv[20]; int li[20];
#pragma unroll
    for (int i = 0; i < 20; ++i) { lv[i] = NEG_INF; li[i] = 0; }

    for (int c0 = 0; c0 < N; c0 += KC) {
        __syncthreads();
        for (int e = tid; e < C * KC; e += BDIM) {
            int c = e >> 7, j = e & (KC - 1);
            colf[c][j] = feat[b * sb + (long)c * sc + (long)(c0 + j) * sn];
        }
        __syncthreads();
        if (tid < KC) {
            float s = 0.f;
            for (int c = 0; c < C; ++c) { float t = colf[c][tid]; s += t * t; }
            colnorm[tid] = s;
        }
        __syncthreads();
#pragma unroll 1
        for (int sub = 0; sub < KC / 32; ++sub) {
            int j = sub * 32 + lane;
            float dot = 0.f;
            for (int c = 0; c < C; ++c) dot += rowf[w][c] * colf[c][j];
            float pd = 2.f * dot - rn - colnorm[j];
            if (pd > lv[19]) {
                lv[19] = pd; li[19] = c0 + j;
#pragma unroll
                for (int t2 = 19; t2 > 0; --t2) {
                    if (lv[t2] > lv[t2 - 1]) {
                        float tv = lv[t2]; lv[t2] = lv[t2 - 1]; lv[t2 - 1] = tv;
                        int   ti = li[t2]; li[t2] = li[t2 - 1]; li[t2 - 1] = ti;
                    }
                }
            }
        }
    }

    const int row = r0 + w;
    for (int kk = 0; kk < 20; ++kk) {
        float v = lv[0]; int who = lane;
        for (int off = 16; off > 0; off >>= 1) {
            float ov = __shfl_xor(v, off, 32);
            int   ow = __shfl_xor(who, off, 32);
            if (ov > v || (ov == v && ow < who)) { v = ov; who = ow; }
        }
        int widx = __shfl(li[0], who, 32);
        if (lane == 0) idxOut[((long)b * N + row) * 20 + kk] = widx;
        if (lane == who) {
#pragma unroll
            for (int t2 = 0; t2 < 19; ++t2) { lv[t2] = lv[t2 + 1]; li[t2] = li[t2 + 1]; }
            lv[19] = NEG_INF;
        }
    }
}

// ---------------------------------------------------------------------------
// WMMA GEMM: Y[b] = act(bn( (sA*Wa + sB*Wb) @ X[b] + ebias ))
// Block tile 64x64, BK=32, 8 waves, each wave: 16(M) x 32(N) -> two wmma accs.
// Branch-free staging: clamped addresses + select; prefetch next X tile.
// ---------------------------------------------------------------------------
__global__ __launch_bounds__(BDIM) void wmma_gemm_kernel(
    const float* __restrict__ Wa, const float* __restrict__ Wb,
    float sA, float sB, int lda,
    int M, int K, int N,
    const float* __restrict__ X, long xs_b, long xs_k, long xs_n,
    float* __restrict__ Y, long ys_b, long ys_m, long ys_n,
    const float* __restrict__ bn_g, const float* __restrict__ bn_b,
    const float* __restrict__ bn_m, const float* __restrict__ bn_v,
    const float* __restrict__ ebias, int do_lrelu)
{
    __shared__ _Float16 As[64][40];   // [m][k]
    __shared__ _Float16 Bs[64][40];   // [n][k]

    const int b   = blockIdx.z;
    const int n0  = blockIdx.x * 64;
    const int m0  = blockIdx.y * 64;
    const int tid = threadIdx.x, lane = tid & 31;
    const int w   = tid >> 5;
    const int msub = (w & 3) * 16;
    const int nsub = (w >> 2) * 32;

    v8f acc0 = {}, acc1 = {};

    for (int k0 = 0; k0 < K; k0 += 32) {
        __syncthreads();
        // stage A (weights), f32 -> f16, branch-free (clamped address + select)
#pragma unroll
        for (int e = tid; e < 64 * 32; e += BDIM) {
            int m = e >> 5, k = e & 31;
            int mc = m0 + m; if (mc > M - 1) mc = M - 1;
            int kc = k0 + k; if (kc > K - 1) kc = K - 1;
            long off = (long)mc * lda + kc;
            float av = sA * Wa[off] + sB * Wb[off];
            bool inb = (m0 + m < M) && (k0 + k < K);
            As[m][k] = (_Float16)(inb ? av : 0.f);
        }
        // stage X tile transposed to [n][k], branch-free; prefetch next tile
#pragma unroll
        for (int e = tid; e < 64 * 32; e += BDIM) {
            int k = e >> 6, n = e & 63;
            int kc = k0 + k; if (kc > K - 1) kc = K - 1;
            float xv = X[b * xs_b + (long)kc * xs_k + (long)(n0 + n) * xs_n];
            Bs[n][k] = (_Float16)((k0 + k < K) ? xv : 0.f);
            if (k0 + 32 < K) {
                int kp = k0 + 32 + k; if (kp > K - 1) kp = K - 1;
                __builtin_prefetch(&X[b * xs_b + (long)kp * xs_k + (long)(n0 + n) * xs_n], 0, 0);
            }
        }
        __syncthreads();

        // A fragment: lane L -> M = L%16; K = j + (L<16?0:16)
        v16h a;
        {
            int row = msub + (lane & 15);
            int kb  = (lane < 16) ? 0 : 16;
#pragma unroll
            for (int j = 0; j < 16; ++j) {
                int kk = kb + j;
                a[j] = As[row][kk];
            }
        }
        // B fragments: lane L -> N = L%16; K = j + (L<16?0:16)
        v16h b0, b1;
        {
            int col = nsub + (lane & 15);
            int kb  = (lane < 16) ? 0 : 16;
#pragma unroll
            for (int j = 0; j < 16; ++j) {
                b0[j] = Bs[col][kb + j];
                b1[j] = Bs[col + 16][kb + j];
            }
        }
        acc0 = __builtin_amdgcn_wmma_f32_16x16x32_f16(false, a, false, b0,
                                                      (short)0, acc0, false, false);
        acc1 = __builtin_amdgcn_wmma_f32_16x16x32_f16(false, a, false, b1,
                                                      (short)0, acc1, false, false);
    }

    // Epilogue: C/D layout — lane L: N = L%16, VGPR v: M = v + (L<16?0:8)
    // N is always a multiple of 64 here; only guard M (final layer M=50).
#pragma unroll
    for (int half = 0; half < 2; ++half) {
        v8f acc = half ? acc1 : acc0;
        int nn = n0 + nsub + half * 16 + (lane & 15);
#pragma unroll
        for (int v = 0; v < 8; ++v) {
            int mm = m0 + msub + v + ((lane < 16) ? 0 : 8);
            if (mm < M) {
                float val = acc[v];
                if (ebias) val += ebias[(long)b * M + mm];
                if (bn_g) {
                    float s = bn_g[mm] * rsqrtf(bn_v[mm] + 1e-5f);
                    val = (val - bn_m[mm]) * s + bn_b[mm];
                }
                if (do_lrelu) val = val > 0.f ? val : 0.2f * val;
                Y[b * ys_b + (long)mm * ys_m + (long)nn * ys_n] = val;
            }
        }
    }
}

// ---------------------------------------------------------------------------
// EdgeConv tail: h[b,o,n] = lrelu(bn( max_k y1[b,o,idx[b,n,k]] + yd[b,o,n] ))
// ---------------------------------------------------------------------------
__global__ __launch_bounds__(BDIM) void edge_max_kernel(
    const float* __restrict__ y1, const float* __restrict__ yd,
    const int* __restrict__ knn, int N,
    const float* __restrict__ g, const float* __restrict__ bb,
    const float* __restrict__ mm, const float* __restrict__ vv,
    float* __restrict__ out, long outStrideB, int chOff)
{
    long t = (long)blockIdx.x * BDIM + threadIdx.x;
    int n = (int)(t % N); long r = t / N;
    int o = (int)(r % 64); int b = (int)(r / 64);

    const int*   ip   = knn + ((long)b * N + n) * 20;
    const float* yrow = y1 + ((long)b * 64 + o) * N;
    float m = NEG_INF;
#pragma unroll
    for (int k = 0; k < 20; ++k) m = fmaxf(m, yrow[ip[k]]);
    float val = m + yd[((long)b * 64 + o) * N + n];
    float s = g[o] * rsqrtf(vv[o] + 1e-5f);
    val = (val - mm[o]) * s + bb[o];
    val = val > 0.f ? val : 0.2f * val;
    out[(long)b * outStrideB + (long)(chOff + o) * N + n] = val;
}

// ---------------------------------------------------------------------------
// Pool: xg[b, ch] = max_n xe, xg[b, 1024+ch] = mean_n xe. One wave per (b,ch).
// ---------------------------------------------------------------------------
__global__ __launch_bounds__(BDIM) void pool_kernel(
    const float* __restrict__ xe, int N, float* __restrict__ xg)
{
    const int tid = threadIdx.x, w = tid >> 5, lane = tid & 31;
    long gidx = (long)blockIdx.x * 8 + w;       // b*1024 + ch
    int b = (int)(gidx / 1024), ch = (int)(gidx % 1024);
    const float* p = xe + gidx * N;
    float mx = NEG_INF, sm = 0.f;
    for (int i = lane; i < N; i += 32) { float v = p[i]; mx = fmaxf(mx, v); sm += v; }
    for (int off = 16; off > 0; off >>= 1) {
        mx = fmaxf(mx, __shfl_xor(mx, off, 32));
        sm += __shfl_xor(sm, off, 32);
    }
    if (lane == 0) {
        xg[(long)b * 2048 + ch]        = mx;
        xg[(long)b * 2048 + 1024 + ch] = sm / (float)N;
    }
}

// ---------------------------------------------------------------------------
// cvec[b,o] = mlp_w1[o, 192:2240] . xg[b, :]   (GEMV, one wave per (b,o))
// ---------------------------------------------------------------------------
__global__ __launch_bounds__(BDIM) void cvec_kernel(
    const float* __restrict__ w1, const float* __restrict__ xg,
    float* __restrict__ cvec)
{
    const int tid = threadIdx.x, w = tid >> 5, lane = tid & 31;
    long gidx = (long)blockIdx.x * 8 + w;       // b*256 + o
    int b = (int)(gidx / 256), o = (int)(gidx % 256);
    const float* wr = w1 + (long)o * 2240 + 192;
    const float* x  = xg + (long)b * 2048;
    float s = 0.f;
    for (int i = lane; i < 2048; i += 32) s += wr[i] * x[i];
    for (int off = 16; off > 0; off >>= 1) s += __shfl_xor(s, off, 32);
    if (lane == 0) cvec[(long)b * 256 + o] = s;
}

// ---------------------------------------------------------------------------
extern "C" void kernel_launch(void* const* d_in, const int* in_sizes, int n_in,
                              void* d_out, int out_size, void* d_ws, size_t ws_size,
                              hipStream_t stream)
{
    const int B = 4, N = 4096;

    const float* x      = (const float*)d_in[0];
    const float* convw[3] = { (const float*)d_in[1], (const float*)d_in[2], (const float*)d_in[3] };
    const float* cbn[3][4];
    for (int l = 0; l < 3; ++l)
        for (int j = 0; j < 4; ++j) cbn[l][j] = (const float*)d_in[4 + l * 4 + j];
    const float* emb_w = (const float*)d_in[16];
    const float* ebn[4] = { (const float*)d_in[17], (const float*)d_in[18],
                            (const float*)d_in[19], (const float*)d_in[20] };
    const float* mlpw[3] = { (const float*)d_in[21], (const float*)d_in[22], (const float*)d_in[23] };
    const float* mbn[3][4];
    for (int l = 0; l < 3; ++l)
        for (int j = 0; j < 4; ++j) mbn[l][j] = (const float*)d_in[24 + l * 4 + j];
    const float* out_w = (const float*)d_in[36];

    // workspace layout (float elements)
    float* ws = (float*)d_ws;
    size_t o_idx  = 0;                                    // B*N*20 ints
    size_t o_y1   = o_idx  + (size_t)B * N * 20;
    size_t o_yd   = o_y1   + (size_t)B * 64 * N;
    size_t o_xcat = o_yd   + (size_t)B * 64 * N;
    size_t o_xe   = o_xcat + (size_t)B * 192 * N;
    size_t o_xg   = o_xe   + (size_t)B * 1024 * N;
    size_t o_cvec = o_xg   + (size_t)B * 2048;
    size_t o_h1   = o_cvec + (size_t)B * 256;
    size_t o_h2   = o_h1   + (size_t)B * 256 * N;
    size_t o_h3   = o_h2   + (size_t)B * 256 * N;

    int*   idxb = (int*)(ws + o_idx);
    float* y1   = ws + o_y1;
    float* yd   = ws + o_yd;
    float* xcat = ws + o_xcat;
    float* xe   = ws + o_xe;
    float* xg   = ws + o_xg;
    float* cvec = ws + o_cvec;
    float* h1   = ws + o_h1;
    float* h2   = ws + o_h2;
    float* h3   = ws + o_h3;

    const dim3 blk(BDIM);
    const dim3 gknn(N / 8, B);

    auto gemm = [&](const float* Wa, const float* Wb, float sA, float sB, int lda,
                    int M, int K,
                    const float* X, long xsb, long xsk, long xsn,
                    float* Y, long ysb, long ysm, long ysn,
                    const float* g, const float* bb_, const float* mm_, const float* vv_,
                    const float* eb, int lrelu) {
        dim3 grid((N + 63) / 64, (M + 63) / 64, B);
        hipLaunchKernelGGL(wmma_gemm_kernel, grid, blk, 0, stream,
                           Wa, Wb, sA, sB, lda, M, K, N, X, xsb, xsk, xsn,
                           Y, ysb, ysm, ysn, g, bb_, mm_, vv_, eb, lrelu);
    };

    // ---------------- EdgeConv layers ----------------
    for (int l = 0; l < 3; ++l) {
        const float* feat; long fsb, fsc, fsn; int C;
        if (l == 0) { feat = x; fsb = (long)N * 3; fsc = 1; fsn = 3; C = 3; }
        else { feat = xcat + (size_t)(l - 1) * 64 * N; fsb = (long)192 * N; fsc = N; fsn = 1; C = 64; }

        hipLaunchKernelGGL(knn_kernel, gknn, blk, 0, stream,
                           feat, fsb, fsc, fsn, C, N, idxb);

        // y1 = W1 @ X ; yd = (W2 - W1) @ X
        gemm(convw[l], convw[l], 1.f, 0.f, 2 * C, 64, C, feat, fsb, fsc, fsn,
             y1, (long)64 * N, N, 1, nullptr, nullptr, nullptr, nullptr, nullptr, 0);
        gemm(convw[l], convw[l] + C, -1.f, 1.f, 2 * C, 64, C, feat, fsb, fsc, fsn,
             yd, (long)64 * N, N, 1, nullptr, nullptr, nullptr, nullptr, nullptr, 0);

        dim3 gem((unsigned)(((long)B * 64 * N) / BDIM));
        hipLaunchKernelGGL(edge_max_kernel, gem, blk, 0, stream,
                           y1, yd, idxb, N,
                           cbn[l][0], cbn[l][1], cbn[l][2], cbn[l][3],
                           xcat, (long)192 * N, l * 64);
    }

    // ---------------- embedding + pooling ----------------
    gemm(emb_w, emb_w, 1.f, 0.f, 192, 1024, 192, xcat, (long)192 * N, N, 1,
         xe, (long)1024 * N, N, 1, ebn[0], ebn[1], ebn[2], ebn[3], nullptr, 1);

    hipLaunchKernelGGL(pool_kernel, dim3(B * 1024 / 8), blk, 0, stream, xe, N, xg);
    hipLaunchKernelGGL(cvec_kernel, dim3(B * 256 / 8), blk, 0, stream, mlpw[0], xg, cvec);

    // ---------------- MLP head ----------------
    gemm(mlpw[0], mlpw[0], 1.f, 0.f, 2240, 256, 192, xcat, (long)192 * N, N, 1,
         h1, (long)256 * N, N, 1, mbn[0][0], mbn[0][1], mbn[0][2], mbn[0][3], cvec, 1);
    gemm(mlpw[1], mlpw[1], 1.f, 0.f, 256, 256, 256, h1, (long)256 * N, N, 1,
         h2, (long)256 * N, N, 1, mbn[1][0], mbn[1][1], mbn[1][2], mbn[1][3], nullptr, 1);
    gemm(mlpw[2], mlpw[2], 1.f, 0.f, 256, 128, 256, h2, (long)256 * N, N, 1,
         h3, (long)128 * N, N, 1, mbn[2][0], mbn[2][1], mbn[2][2], mbn[2][3], nullptr, 1);

    // out: (50,128) @ h3, no bn/act, stored transposed as (B, N, 50)
    gemm(out_w, out_w, 1.f, 0.f, 128, 50, 128, h3, (long)128 * N, N, 1,
         (float*)d_out, (long)N * 50, 1, 50, nullptr, nullptr, nullptr, nullptr, nullptr, 0);
}
